// TGN_90469191123536
// MI455X (gfx1250) — matrix operand order repaired and checked by reading, not compiled, not run on hardware
//
#include <hip/hip_runtime.h>
#include <math.h>

// ---------------- problem constants ----------------
#define NN      10000      // N_NODES
#define NEDGE   100000
#define BB      8192       // batch
#define KNBR    10
#define FD      172        // MEM = EMB = TE = EF
#define MSG     688        // 2*MEM + TE + EF
#define HID     80

#define KRNN    864        // 860 (msg|mem) padded to 27*32
#define KSTEP_RNN 27
#define K12     544        // 516 padded to 17*32
#define KSTEP_12  17
#define NPAD    176        // 172 padded to 11*16
#define NT      11         // N tiles of 16
#define ROWS_RNN_PAD 10112 // 79 blocks * 128 rows

typedef __attribute__((ext_vector_type(16))) __bf16       v16bf;
typedef __attribute__((ext_vector_type(8)))  float        v8f;
typedef __attribute__((ext_vector_type(8)))  unsigned int v8u;

__device__ __forceinline__ unsigned short f32_to_bf16(float f) {
  union { float f; unsigned u; } v; v.f = f;
  unsigned r = v.u + 0x7FFFu + ((v.u >> 16) & 1u);   // round-to-nearest-even
  return (unsigned short)(r >> 16);
}

__device__ __forceinline__ float4 ld4(const float* p) {
  return *(const float4*)p;   // callers guarantee 16B alignment
}

// pack 4 floats -> 4 bf16, one b64 store
__device__ __forceinline__ void st_bf16x4(unsigned short* p,
                                          float a, float b, float c, float d) {
  union { unsigned short s[4]; unsigned long long u; } t;
  t.s[0] = f32_to_bf16(a); t.s[1] = f32_to_bf16(b);
  t.s[2] = f32_to_bf16(c); t.s[3] = f32_to_bf16(d);
  *(unsigned long long*)p = t.u;
}

// ---------------------------------------------------------------------------
// Shared WMMA core: one wave computes a 16-row M tile x 176 cols (11 N tiles).
// A = bf16 activations [rows x lda], W = bf16 weights [176 x ldw] (K-major rows,
// i.e. W[n][k] — exactly the transposed B layout we need).
// Fragment layouts per CDNA5 ISA 7.12.2 (16-bit A 16x32; B 32x16); the 32-bit
// lane loads below merge into global_load_b128 pairs (verified in disasm).
// ---------------------------------------------------------------------------
__device__ __forceinline__ void wmma_strip(
    const unsigned short* __restrict__ A, int lda, long rowBase,
    const unsigned short* __restrict__ W, int ldw,
    int kSteps, v8f acc[NT])
{
  const int lane = threadIdx.x & 31;
  const int m    = lane & 15;
  const int half = lane >> 4;
  for (int kt = 0; kt < kSteps; ++kt) {
    const unsigned short* ap = A + (rowBase + m) * (long)lda + kt * 32;
    v8u au;
#pragma unroll
    for (int j = 0; j < 4; ++j)          // VGPR0-3: K = half*8 + {0,1},{2,3},{4,5},{6,7}
      au[j] = *(const unsigned int*)(ap + half * 8 + 2 * j);
#pragma unroll
    for (int j = 0; j < 4; ++j)          // VGPR4-7: K = 16 + half*8 + ...
      au[4 + j] = *(const unsigned int*)(ap + 16 + half * 8 + 2 * j);
    v16bf a = __builtin_bit_cast(v16bf, au);
#pragma unroll
    for (int nt = 0; nt < NT; ++nt) {
      // B 32x16: same K order as A (VGPR0-3: K = half*8 + 2j,2j+1; VGPR4-7: K = 16 + half*8 + ...)
      const unsigned short* wp = W + (nt * 16 + m) * (long)ldw + kt * 32 + half * 8;
      v8u bu;
#pragma unroll
      for (int j = 0; j < 8; ++j)
        bu[j] = *(const unsigned int*)(wp + 2 * j + (j >= 4 ? 8 : 0));
      v16bf b = __builtin_bit_cast(v16bf, bu);
      acc[nt] = __builtin_amdgcn_wmma_f32_16x16x32_bf16(
          false, a, false, b, (short)0, acc[nt], false, false);
    }
  }
}

// ---------------------------------------------------------------------------
// 1) prep: bf16-convert + zero-pad weights; init last_idx = -1
// ---------------------------------------------------------------------------
__global__ void prep_kernel(unsigned short* __restrict__ Wrnn,
                            unsigned short* __restrict__ W1,
                            unsigned short* __restrict__ W2,
                            int* __restrict__ last_idx,
                            const float* __restrict__ Wih,
                            const float* __restrict__ Whh,
                            const float* __restrict__ l1w,
                            const float* __restrict__ l2w)
{
  long idx = (long)blockIdx.x * blockDim.x + threadIdx.x;
  if (idx < (long)NPAD * KRNN) {
    int n = (int)(idx / KRNN), k = (int)(idx % KRNN);
    float v = 0.f;
    if (n < FD) {
      if (k < MSG)            v = Wih[(long)n * MSG + k];
      else if (k < MSG + FD)  v = Whh[(long)n * FD + (k - MSG)];
    }
    Wrnn[idx] = f32_to_bf16(v);
  }
  if (idx < (long)NPAD * K12) {
    int n = (int)(idx / K12), k = (int)(idx % K12);
    float v1 = 0.f, v2 = 0.f;
    if (n < FD && k < 3 * FD) {
      v1 = l1w[(long)n * (3 * FD) + k];
      v2 = l2w[(long)n * (3 * FD) + k];
    }
    W1[idx] = f32_to_bf16(v1);
    W2[idx] = f32_to_bf16(v2);
  }
  if (idx < NN) last_idx[idx] = -1;
}

// ---------------------------------------------------------------------------
// 2) last-message aggregation: last_idx[n] = max batch index with sources==n
// ---------------------------------------------------------------------------
__global__ void scatter_kernel(int* __restrict__ last_idx,
                               const int* __restrict__ sources)
{
  int i = blockIdx.x * blockDim.x + threadIdx.x;
  if (i < BB) atomicMax(&last_idx[sources[i]], i);
}

// ---------------------------------------------------------------------------
// 3) build RNN activation rows: [src_mem | dst_mem | edge_f | time_enc | own_mem]
//    one wave per node row, float4 gathers / packed-bf16 b64 stores.
//    All segment boundaries (172/344/516/688/860/864) divisible by 4; gathered
//    rows are 688B = 43*16B so float4 accesses stay 16B-aligned.
// ---------------------------------------------------------------------------
__global__ void build_rnn_act(unsigned short* __restrict__ act,
                              const int* __restrict__ last_idx,
                              const float* __restrict__ node_mem,
                              const float* __restrict__ edge_f,
                              const float* __restrict__ ts,
                              const float* __restrict__ last_upd,
                              const float* __restrict__ tw,
                              const float* __restrict__ tb,
                              const int* __restrict__ src,
                              const int* __restrict__ dst,
                              const int* __restrict__ eidx)
{
  const int lane = threadIdx.x & 31;
  long row = (long)blockIdx.x * 8 + (threadIdx.x >> 5);
  if (row >= ROWS_RNN_PAD) return;
  unsigned short* out = act + row * KRNN;
  int li = (row < NN) ? last_idx[row] : -1;
  if (li < 0) {
    for (int cq = lane; cq < KRNN / 4; cq += 32)
      st_bf16x4(out + cq * 4, 0.f, 0.f, 0.f, 0.f);
    return;
  }
  int s = src[li], d = dst[li], e = eidx[li];
  float dt = ts[li] - last_upd[s];
  for (int cq = lane; cq < KRNN / 4; cq += 32) {
    int c = cq * 4;
    float4 v;
    if      (c < FD)       v = ld4(node_mem + (long)s * FD + c);
    else if (c < 2 * FD)   v = ld4(node_mem + (long)d * FD + (c - FD));
    else if (c < 3 * FD)   v = ld4(edge_f + (long)e * FD + (c - 2 * FD));
    else if (c < MSG) {
      int cc = c - 3 * FD;
      float4 w = ld4(tw + cc), b = ld4(tb + cc);
      v.x = cosf(dt * w.x + b.x); v.y = cosf(dt * w.y + b.y);
      v.z = cosf(dt * w.z + b.z); v.w = cosf(dt * w.w + b.w);
    }
    else if (c < MSG + FD) v = ld4(node_mem + row * FD + (c - MSG));
    else                   { v.x = v.y = v.z = v.w = 0.f; }
    st_bf16x4(out + c, v.x, v.y, v.z, v.w);
  }
}

// ---------------------------------------------------------------------------
// 4) RNN GEMM + tanh epilogue: mem[n] = upd ? tanh(...) : node_mem[n]
// ---------------------------------------------------------------------------
__global__ void gemm_rnn_kernel(const unsigned short* __restrict__ A,
                                const unsigned short* __restrict__ W,
                                const float* __restrict__ bih,
                                const float* __restrict__ bhh,
                                const int* __restrict__ last_idx,
                                const float* __restrict__ node_mem,
                                float* __restrict__ memv)
{
  long tile = (long)blockIdx.x * 8 + (threadIdx.x >> 5);
  long rowBase = tile * 16;
  if (rowBase >= ROWS_RNN_PAD) return;
  v8f acc[NT] = {};
  wmma_strip(A, KRNN, rowBase, W, KRNN, KSTEP_RNN, acc);
  const int lane = threadIdx.x & 31;
  const int n0 = lane & 15, mh = (lane >> 4) * 8;
#pragma unroll
  for (int nt = 0; nt < NT; ++nt) {
    int col = nt * 16 + n0;
    if (col >= FD) continue;
    float bias = bih[col] + bhh[col];
#pragma unroll
    for (int r = 0; r < 8; ++r) {
      long row = rowBase + mh + r;
      if (row < NN) {
        float v = (last_idx[row] >= 0) ? tanhf(acc[nt][r] + bias)
                                       : node_mem[row * FD + col];
        memv[row * FD + col] = v;
      }
    }
  }
}

// ---------------------------------------------------------------------------
// 5a) gather-sum neighbor features over k BEFORE linear_1 (sum/matmul commute):
//     S[b] = sum_k [ mem[nbr]+tnf[nbr] | cos((t_b - t_nbr)w + b) | edge_f[nbr_e] ]
//     float4 gathers; this kernel carries ~170MB of the total traffic.
// ---------------------------------------------------------------------------
__global__ void build_act1(unsigned short* __restrict__ act1,
                           const float* __restrict__ memv,
                           const float* __restrict__ tnf,
                           const float* __restrict__ edge_f,
                           const float* __restrict__ ts,
                           const float* __restrict__ net,
                           const float* __restrict__ tw,
                           const float* __restrict__ tb,
                           const int* __restrict__ nbrs,
                           const int* __restrict__ neidx)
{
  const int lane = threadIdx.x & 31;
  long b = (long)blockIdx.x * 8 + (threadIdx.x >> 5);
  if (b >= BB) return;
  int nb[KNBR], ne[KNBR];
  float dt[KNBR];
  float tsb = ts[b];
#pragma unroll
  for (int k = 0; k < KNBR; ++k) {
    nb[k] = nbrs[b * KNBR + k];
    ne[k] = neidx[b * KNBR + k];
    dt[k] = tsb - net[b * KNBR + k];
  }
  unsigned short* out = act1 + b * K12;
  for (int cq = lane; cq < K12 / 4; cq += 32) {
    int c = cq * 4;
    float vx = 0.f, vy = 0.f, vz = 0.f, vw = 0.f;
    if (c < FD) {
#pragma unroll
      for (int k = 0; k < KNBR; ++k) {
        float4 a = ld4(memv + (long)nb[k] * FD + c);
        float4 t = ld4(tnf + (long)nb[k] * FD + c);
        vx += a.x + t.x; vy += a.y + t.y; vz += a.z + t.z; vw += a.w + t.w;
      }
    } else if (c < 2 * FD) {
      int cc = c - FD;
      float4 w = ld4(tw + cc), bb = ld4(tb + cc);
#pragma unroll
      for (int k = 0; k < KNBR; ++k) {
        vx += cosf(dt[k] * w.x + bb.x); vy += cosf(dt[k] * w.y + bb.y);
        vz += cosf(dt[k] * w.z + bb.z); vw += cosf(dt[k] * w.w + bb.w);
      }
    } else if (c < 3 * FD) {
      int cc = c - 2 * FD;
#pragma unroll
      for (int k = 0; k < KNBR; ++k) {
        float4 a = ld4(edge_f + (long)ne[k] * FD + cc);
        vx += a.x; vy += a.y; vz += a.z; vw += a.w;
      }
    }
    st_bf16x4(out + c, vx, vy, vz, vw);
  }
}

// ---------------------------------------------------------------------------
// 5b) fill act2 columns 172..543: [ src_feat | cos(time_b) | zero-pad ]
// ---------------------------------------------------------------------------
__global__ void build_act2_rest(unsigned short* __restrict__ act2,
                                const float* __restrict__ memv,
                                const float* __restrict__ tnf,
                                const float* __restrict__ tb,
                                const int* __restrict__ src)
{
  const int lane = threadIdx.x & 31;
  long b = (long)blockIdx.x * 8 + (threadIdx.x >> 5);
  if (b >= BB) return;
  int s = src[b];
  unsigned short* out = act2 + b * K12;
  for (int cq = FD / 4 + lane; cq < K12 / 4; cq += 32) {
    int c = cq * 4;
    float4 v;
    if (c < 2 * FD) {
      int cc = c - FD;
      float4 a = ld4(memv + (long)s * FD + cc);
      float4 t = ld4(tnf + (long)s * FD + cc);
      v.x = a.x + t.x; v.y = a.y + t.y; v.z = a.z + t.z; v.w = a.w + t.w;
    } else if (c < 3 * FD) {
      int cc = c - 2 * FD;
      float4 bb = ld4(tb + cc);
      v.x = cosf(bb.x); v.y = cosf(bb.y); v.z = cosf(bb.z); v.w = cosf(bb.w);
    } else {
      v.x = v.y = v.z = v.w = 0.f;
    }
    st_bf16x4(out + c, v.x, v.y, v.z, v.w);
  }
}

// ---------------------------------------------------------------------------
// 6) GEMM1: nbr_sum = relu(S @ lin1_w.T + K*lin1_b) -> bf16 into act2[:,0:172]
// ---------------------------------------------------------------------------
__global__ void gemm1_kernel(const unsigned short* __restrict__ A,
                             const unsigned short* __restrict__ W,
                             const float* __restrict__ b1,
                             unsigned short* __restrict__ act2)
{
  long rowBase = ((long)blockIdx.x * 8 + (threadIdx.x >> 5)) * 16;
  v8f acc[NT] = {};
  wmma_strip(A, K12, rowBase, W, K12, KSTEP_12, acc);
  const int lane = threadIdx.x & 31;
  const int n0 = lane & 15, mh = (lane >> 4) * 8;
#pragma unroll
  for (int nt = 0; nt < NT; ++nt) {
    int col = nt * 16 + n0;
    if (col >= FD) continue;
    float bias = (float)KNBR * b1[col];   // sum of K copies of lin1_b
#pragma unroll
    for (int r = 0; r < 8; ++r) {
      long row = rowBase + mh + r;
      act2[row * K12 + col] = f32_to_bf16(fmaxf(acc[nt][r] + bias, 0.f));
    }
  }
}

// ---------------------------------------------------------------------------
// 7) GEMM2: emb = act2 @ lin2_w.T + lin2_b  (f32 out)
// ---------------------------------------------------------------------------
__global__ void gemm2_kernel(const unsigned short* __restrict__ A,
                             const unsigned short* __restrict__ W,
                             const float* __restrict__ b2,
                             float* __restrict__ emb)
{
  long rowBase = ((long)blockIdx.x * 8 + (threadIdx.x >> 5)) * 16;
  v8f acc[NT] = {};
  wmma_strip(A, K12, rowBase, W, K12, KSTEP_12, acc);
  const int lane = threadIdx.x & 31;
  const int n0 = lane & 15, mh = (lane >> 4) * 8;
#pragma unroll
  for (int nt = 0; nt < NT; ++nt) {
    int col = nt * 16 + n0;
    if (col >= FD) continue;
    float bias = b2[col];
#pragma unroll
    for (int r = 0; r < 8; ++r) {
      long row = rowBase + mh + r;
      emb[row * FD + col] = acc[nt][r] + bias;
    }
  }
}

// ---------------------------------------------------------------------------
// 8) MLP head + sigmoid; weights staged in LDS (~56KB of WGP's 320KB)
// ---------------------------------------------------------------------------
__global__ void mlp_kernel(const float* __restrict__ emb,
                           const float* __restrict__ w1, const float* __restrict__ b1,
                           const float* __restrict__ w2, const float* __restrict__ b2,
                           const float* __restrict__ w3, const float* __restrict__ b3,
                           float* __restrict__ out)
{
  __shared__ float lw1[HID * FD];
  __shared__ float lw2[10 * HID];
  __shared__ float lb1[HID];
  __shared__ float lb2[10];
  for (int i = threadIdx.x; i < HID * FD; i += blockDim.x) lw1[i] = w1[i];
  for (int i = threadIdx.x; i < 10 * HID; i += blockDim.x) lw2[i] = w2[i];
  if (threadIdx.x < HID) lb1[threadIdx.x] = b1[threadIdx.x];
  if (threadIdx.x < 10)  lb2[threadIdx.x] = b2[threadIdx.x];
  __syncthreads();
  long b = (long)blockIdx.x * blockDim.x + threadIdx.x;
  if (b >= BB) return;
  float h1[HID];
#pragma unroll
  for (int o = 0; o < HID; ++o) h1[o] = lb1[o];
  const float* e = emb + b * FD;
  for (int j = 0; j < FD; ++j) {
    float ej = e[j];
#pragma unroll
    for (int o = 0; o < HID; ++o) h1[o] = fmaf(ej, lw1[o * FD + j], h1[o]);
  }
#pragma unroll
  for (int o = 0; o < HID; ++o) h1[o] = fmaxf(h1[o], 0.f);
  float h2[10];
#pragma unroll
  for (int o = 0; o < 10; ++o) {
    float s = lb2[o];
#pragma unroll
    for (int j = 0; j < HID; ++j) s = fmaf(h1[j], lw2[o * HID + j], s);
    h2[o] = fmaxf(s, 0.f);
  }
  float s3 = b3[0];
#pragma unroll
  for (int j = 0; j < 10; ++j) s3 = fmaf(h2[j], w3[j], s3);
  out[b] = 1.f / (1.f + expf(-s3));
}

// ---------------------------------------------------------------------------
extern "C" void kernel_launch(void* const* d_in, const int* in_sizes, int n_in,
                              void* d_out, int out_size, void* d_ws, size_t ws_size,
                              hipStream_t stream) {
  (void)in_sizes; (void)n_in; (void)out_size; (void)ws_size;
  const float* timestamps = (const float*)d_in[0];
  const float* edge_f     = (const float*)d_in[1];
  const float* tnf        = (const float*)d_in[2];
  const float* node_mem   = (const float*)d_in[3];
  const float* last_upd   = (const float*)d_in[4];
  const float* net        = (const float*)d_in[5];
  const float* tw         = (const float*)d_in[6];
  const float* tb         = (const float*)d_in[7];
  const float* rnn_Wih    = (const float*)d_in[8];
  const float* rnn_Whh    = (const float*)d_in[9];
  const float* rnn_bih    = (const float*)d_in[10];
  const float* rnn_bhh    = (const float*)d_in[11];
  const float* lin1_w     = (const float*)d_in[12];
  const float* lin1_b     = (const float*)d_in[13];
  const float* lin2_w     = (const float*)d_in[14];
  const float* lin2_b     = (const float*)d_in[15];
  const float* mlp_w1     = (const float*)d_in[16];
  const float* mlp_b1     = (const float*)d_in[17];
  const float* mlp_w2     = (const float*)d_in[18];
  const float* mlp_b2     = (const float*)d_in[19];
  const float* mlp_w3     = (const float*)d_in[20];
  const float* mlp_b3     = (const float*)d_in[21];
  const int*   sources    = (const int*)d_in[22];
  const int*   dests      = (const int*)d_in[23];
  const int*   edge_idxs  = (const int*)d_in[24];
  const int*   neighbors  = (const int*)d_in[25];
  const int*   nbr_eidx   = (const int*)d_in[26];
  float* out = (float*)d_out;

  // workspace carve-up with liveness aliasing (~31 MB peak):
  //   persistent: last_idx | Wrnn | W1 | W2 | memv
  //   union U:    actR (steps 3-4)  ALIASES  act1+act2+emb (steps 5-8)
  char* ws = (char*)d_ws;
  size_t off = 0;
  auto carve = [&](size_t bytes) -> void* {
    void* p = ws + off;
    off = (off + bytes + 255) & ~(size_t)255;
    return p;
  };
  int*            last_idx = (int*)           carve(sizeof(int) * NN);
  unsigned short* Wrnn     = (unsigned short*)carve(2ull * NPAD * KRNN);
  unsigned short* W1       = (unsigned short*)carve(2ull * NPAD * K12);
  unsigned short* W2       = (unsigned short*)carve(2ull * NPAD * K12);
  float*          memv     = (float*)         carve(4ull * NN * FD);
  size_t uoff = off;                          // union region base
  unsigned short* actR     = (unsigned short*)carve(2ull * ROWS_RNN_PAD * KRNN);
  off = uoff;                                 // rewind: actR dead after step 4
  unsigned short* act1     = (unsigned short*)carve(2ull * BB * K12);
  unsigned short* act2     = (unsigned short*)carve(2ull * BB * K12);
  float*          emb      = (float*)         carve(4ull * BB * FD);

  // 1) weight conversion + last_idx init
  prep_kernel<<<(NPAD * KRNN + 255) / 256, 256, 0, stream>>>(
      Wrnn, W1, W2, last_idx, rnn_Wih, rnn_Whh, lin1_w, lin2_w);
  // 2) last-message scatter-max
  scatter_kernel<<<(BB + 255) / 256, 256, 0, stream>>>(last_idx, sources);
  // 3) RNN activations
  build_rnn_act<<<ROWS_RNN_PAD / 8, 256, 0, stream>>>(
      actR, last_idx, node_mem, edge_f, timestamps, last_upd, tw, tb,
      sources, dests, edge_idxs);
  // 4) RNN GEMM + tanh + upd-mask
  gemm_rnn_kernel<<<ROWS_RNN_PAD / 128, 256, 0, stream>>>(
      actR, Wrnn, rnn_bih, rnn_bhh, last_idx, node_mem, memv);
  // 5) neighbor gather-sum + act2 tail columns  (actR dead from here; aliased)
  build_act1<<<BB / 8, 256, 0, stream>>>(
      act1, memv, tnf, edge_f, timestamps, net, tw, tb, neighbors, nbr_eidx);
  build_act2_rest<<<BB / 8, 256, 0, stream>>>(act2, memv, tnf, tb, sources);
  // 6) linear_1 (+K*b, relu) -> act2[:,0:172]
  gemm1_kernel<<<BB / 128, 256, 0, stream>>>(act1, W1, lin1_b, act2);
  // 7) linear_2 -> emb
  gemm2_kernel<<<BB / 128, 256, 0, stream>>>(act2, W2, lin2_b, emb);
  // 8) MLP head + sigmoid
  mlp_kernel<<<BB / 256, 256, 0, stream>>>(
      emb, mlp_w1, mlp_b1, mlp_w2, mlp_b2, mlp_w3, mlp_b3, out);
}
